// CrossAttention_19129784336597
// MI455X (gfx1250) — matrix operand order repaired and checked by reading, not compiled, not run on hardware
//
#include <hip/hip_runtime.h>

// ---------------------------------------------------------------------------
// CDNA5 (gfx1250) cross-attention: all matmuls via v_wmma_f32_16x16x32_f16
// Round 4: peeled pipeline epilogues (branch-free steady state), rcp for the
// softmax normalizer. Structure: double-buffered LDS staging feeding WMMAs.
// ---------------------------------------------------------------------------

typedef __attribute__((ext_vector_type(8)))  _Float16 v8h;
typedef __attribute__((ext_vector_type(16))) _Float16 v16h;
typedef __attribute__((ext_vector_type(8)))  float    v8f;

union H16 { v16h v; v8h h[2]; };

// Fragment layout (wave32, 16x16x32 f16 WMMA):
//  A (16x32): lane L -> row M = L&15, g = L>>4.
//    elems 0..7  = K (8g..8g+7), elems 8..15 = K (16+8g..16+8g+7)  (table order)
//  B (32x16): lane L -> col N = L&15, g = L>>4.
//    elems 0..7 = K (8g..8g+7), elems 8..15 = K (16+8g..16+8g+7)  (table order)
//  C/D (16x16 f32): VGPR r, lane L -> row M = r + 8*(L>>4), col N = L&15.

#define NDIM 1024
#define BM 128
#define BN 128
#define BK 32
#define AKV 64  // attention kv tile

// ---------------------------------------------------------------------------
// Tiled GEMM: C[M x 1024] = A[M x Kdim] * W[Kdim x 1024] (+ bias)
// 256 threads = 8 waves (4x2); each wave owns a 32x64 C tile (8 WMMAs/K-step).
// Double-buffered LDS, last K-step peeled for a branch-free main loop.
// ---------------------------------------------------------------------------
template <typename AT, typename WT, typename CT>
__global__ __launch_bounds__(256) void gemm_wmma_k(
    const AT* __restrict__ A, const WT* __restrict__ W,
    const float* __restrict__ bias, CT* __restrict__ C,
    int M, int Kdim) {
  __shared__ __align__(16) _Float16 sA[2][BM][BK + 8];  // A tile, row-major
  __shared__ __align__(16) _Float16 sW[2][BN][BK + 8];  // W tile, TRANSPOSED [n][k]

  const int t    = threadIdx.x;
  const int lane = t & 31;
  const int wave = t >> 5;
  const int wm   = wave >> 1;      // 0..3 (rows)
  const int wn   = wave & 1;       // 0..1 (cols)
  const int lg   = lane >> 4;
  const int ln   = lane & 15;
  const int m0   = blockIdx.y * BM;
  const int n0   = blockIdx.x * BN;

  AT ra[16];
  WT rw[16];
  const int rA = t >> 1;             // A tile: row, 16-elem half-row
  const int cA = (t & 1) * 16;
  const int kW = t & 31;             // W tile: k row, 16-elem col chunk
  const int cW = (t >> 5) * 16;

  auto loadG = [&](int k0) {
    const AT* srcA = A + (size_t)(m0 + rA) * Kdim + k0 + cA;
#pragma unroll
    for (int j = 0; j < 16; ++j) ra[j] = srcA[j];
    const WT* srcW = W + (size_t)(k0 + kW) * NDIM + n0 + cW;
#pragma unroll
    for (int j = 0; j < 16; ++j) rw[j] = srcW[j];
  };
  auto storeL = [&](int buf) {
#pragma unroll
    for (int j = 0; j < 16; ++j) sA[buf][rA][cA + j] = (_Float16)(float)ra[j];
#pragma unroll
    for (int j = 0; j < 16; ++j) sW[buf][cW + j][kW] = (_Float16)(float)rw[j];
  };

  v8f acc[2][4];
#pragma unroll
  for (int i = 0; i < 2; ++i)
#pragma unroll
    for (int j = 0; j < 4; ++j)
#pragma unroll
      for (int e = 0; e < 8; ++e) acc[i][j][e] = 0.f;

  auto compute = [&](int buf) {
    H16 a[2], b[4];
#pragma unroll
    for (int i = 0; i < 2; ++i) {
      const _Float16* ap = &sA[buf][wm * 32 + i * 16 + ln][0];
      a[i].h[0] = *(const v8h*)(ap + 8 * lg);
      a[i].h[1] = *(const v8h*)(ap + 16 + 8 * lg);
    }
#pragma unroll
    for (int j = 0; j < 4; ++j) {
      const _Float16* bp = &sW[buf][wn * 64 + j * 16 + ln][0];
      b[j].h[0] = *(const v8h*)(bp + 8 * lg);
      b[j].h[1] = *(const v8h*)(bp + 16 + 8 * lg);
    }
#pragma unroll
    for (int i = 0; i < 2; ++i)
#pragma unroll
      for (int j = 0; j < 4; ++j)
        acc[i][j] = __builtin_amdgcn_wmma_f32_16x16x32_f16(
            false, a[i].v, false, b[j].v, (short)0, acc[i][j], false, false);
  };

  loadG(0);
  storeL(0);
  __syncthreads();

  int cur = 0;
#pragma unroll 1
  for (int k0 = 0; k0 < Kdim - BK; k0 += BK) {  // branch-free steady state
    loadG(k0 + BK);       // global prefetch in flight during WMMAs
    compute(cur);
    storeL(cur ^ 1);      // other buffer: no reader this iteration
    __syncthreads();
    cur ^= 1;
  }
  compute(cur);           // peeled last K-step

  // Epilogue: C-frag row M = r + 8*lg, col N = ln
#pragma unroll
  for (int i = 0; i < 2; ++i) {
#pragma unroll
    for (int j = 0; j < 4; ++j) {
      const int gn = n0 + wn * 64 + j * 16 + ln;
      const float bv = bias ? bias[gn] : 0.f;
#pragma unroll
      for (int r = 0; r < 8; ++r) {
        const int gm = m0 + wm * 32 + i * 16 + r + 8 * lg;
        C[(size_t)gm * NDIM + gn] = (CT)(acc[i][j][r] + bv);
      }
    }
  }
}

// ---------------------------------------------------------------------------
// Attention softmax: P = softmax(Q K^T / 32) as f16.
// Block = 4 waves; each wave owns a 16-row Q tile, all share the kv loop.
// K tiles [64 kv][32 d] staged cooperatively in double-buffered LDS; Q A-frags
// double-buffered in registers from global (L2 resident). Last d-step peeled.
// Pass 0: online row max / sum-exp stats.  Pass 1: emit normalized P.
// ---------------------------------------------------------------------------
__global__ __launch_bounds__(128) void attn_softmax_k(
    const _Float16* __restrict__ Qh, const _Float16* __restrict__ Kh,
    _Float16* __restrict__ P) {
  __shared__ __align__(16) _Float16 sK[2][AKV][BK + 8];  // 2 x 64 x 40 x 2B

  const int t     = threadIdx.x;
  const int lane  = t & 31;
  const int wave  = t >> 5;                 // 0..3
  const int lg    = lane >> 4;
  const int ln    = lane & 15;
  const int qtile = blockIdx.x * 4 + wave;  // 0..511
  const int b     = qtile >> 7;             // 128 tiles per batch
  const int qr    = (qtile & 127) * 16;

  const _Float16* Qbase  = Qh + ((size_t)b * 2048 + qr) * 1024;
  const _Float16* Kbase  = Kh + (size_t)b * 2048 * 1024;
  _Float16*       Pbase  = P  + ((size_t)b * 2048 + qr) * 2048;
  const _Float16* qrow_p = Qbase + (size_t)ln * 1024;  // A-frag row = lane&15
  const float scale = 0.03125f;                        // 1/sqrt(1024)

  // Cooperative K staging: thread -> kv row t>>1, 16-elem half-row
  const int rK = t >> 1;
  const int cK = (t & 1) * 16;
  v8h kst0, kst1;

  auto loadKg = [&](int kv0, int d0) {
    const _Float16* src = Kbase + (size_t)(kv0 + rK) * 1024 + d0 + cK;
    kst0 = *(const v8h*)src;
    kst1 = *(const v8h*)(src + 8);
  };
  auto storeKl = [&](int buf) {
    *(v8h*)&sK[buf][rK][cK]     = kst0;
    *(v8h*)&sK[buf][rK][cK + 8] = kst1;
  };

  float mrow[8], lrow[8];
#pragma unroll
  for (int r = 0; r < 8; ++r) { mrow[r] = -1e30f; lrow[r] = 0.f; }

  for (int pass = 0; pass < 2; ++pass) {
#pragma unroll 1
    for (int kv0 = 0; kv0 < 2048; kv0 += AKV) {
      v8f s[4];
#pragma unroll
      for (int j = 0; j < 4; ++j)
#pragma unroll
        for (int e = 0; e < 8; ++e) s[j][e] = 0.f;

      auto compute = [&](int buf, const H16& af) {
        H16 bf[4];
#pragma unroll
        for (int j = 0; j < 4; ++j) {
          const _Float16* bp = &sK[buf][j * 16 + ln][0];
          bf[j].h[0] = *(const v8h*)(bp + 8 * lg);
          bf[j].h[1] = *(const v8h*)(bp + 16 + 8 * lg);
        }
#pragma unroll
        for (int j = 0; j < 4; ++j)
          s[j] = __builtin_amdgcn_wmma_f32_16x16x32_f16(
              false, af.v, false, bf[j].v, (short)0, s[j], false, false);
      };

      H16 a0, a1;
      loadKg(kv0, 0);
      storeKl(0);
      a0.h[0] = *(const v8h*)(qrow_p + 8 * lg);
      a0.h[1] = *(const v8h*)(qrow_p + 16 + 8 * lg);
      __syncthreads();

      int cur = 0;
#pragma unroll 1
      for (int d0 = 0; d0 < 1024 - 32; d0 += 32) {  // branch-free steady state
        loadKg(kv0, d0 + 32);  // global prefetch overlaps WMMAs below
        a1.h[0] = *(const v8h*)(qrow_p + d0 + 32 + 8 * lg);
        a1.h[1] = *(const v8h*)(qrow_p + d0 + 48 + 8 * lg);
        compute(cur, a0);
        storeKl(cur ^ 1);
        a0 = a1;
        __syncthreads();
        cur ^= 1;
      }
      compute(cur, a0);  // peeled last d-step (reads buffer written last iter)

      if (pass == 0) {
#pragma unroll
        for (int r = 0; r < 8; ++r) {
          float v = fmaxf(fmaxf(s[0][r], s[1][r]), fmaxf(s[2][r], s[3][r])) * scale;
#pragma unroll
          for (int off = 1; off < 16; off <<= 1)
            v = fmaxf(v, __shfl_xor(v, off, 32));
          const float mnew = fmaxf(mrow[r], v);
          float e = 0.f;
#pragma unroll
          for (int j = 0; j < 4; ++j) e += __expf(s[j][r] * scale - mnew);
#pragma unroll
          for (int off = 1; off < 16; off <<= 1) e += __shfl_xor(e, off, 32);
          lrow[r] = lrow[r] * __expf(mrow[r] - mnew) + e;
          mrow[r] = mnew;
        }
      } else {
#pragma unroll
        for (int r = 0; r < 8; ++r) {
          const float inv = __builtin_amdgcn_rcpf(lrow[r]);  // v_rcp_f32
          const int gm = r + 8 * lg;
#pragma unroll
          for (int j = 0; j < 4; ++j) {
            const float p = __expf(s[j][r] * scale - mrow[r]) * inv;
            Pbase[(size_t)gm * 2048 + kv0 + 16 * j + ln] = (_Float16)p;
          }
        }
      }
    }
  }
}

// ---------------------------------------------------------------------------
// Launch: Q/K/V projections -> softmax(P) -> O = P*V -> out = O*Wo + bo
// Workspace: Qh(16M) Kh(16M) Vh(16M) P(32M) Oh(16M) f16  ~= 96 MB
// ---------------------------------------------------------------------------
extern "C" void kernel_launch(void* const* d_in, const int* in_sizes, int n_in,
                              void* d_out, int out_size, void* d_ws,
                              size_t ws_size, hipStream_t stream) {
  const float* query = (const float*)d_in[0];
  const float* key   = (const float*)d_in[1];
  const float* value = (const float*)d_in[2];
  const float* Wq = (const float*)d_in[3];
  const float* bq = (const float*)d_in[4];
  const float* Wk = (const float*)d_in[5];
  const float* bk = (const float*)d_in[6];
  const float* Wv = (const float*)d_in[7];
  const float* bv = (const float*)d_in[8];
  const float* Wo = (const float*)d_in[9];
  const float* bo = (const float*)d_in[10];
  float* out = (float*)d_out;

  _Float16* Qh = (_Float16*)d_ws;
  _Float16* Kh = Qh + (size_t)8192 * 1024;
  _Float16* Vh = Kh + (size_t)8192 * 1024;
  _Float16* P  = Vh + (size_t)8192 * 1024;
  _Float16* Oh = P  + (size_t)4 * 2048 * 2048;

  const dim3 blk(256);
  // Projections: [8192 x 1024] x [1024 x 1024]
  gemm_wmma_k<float, float, _Float16>
      <<<dim3(NDIM / BN, 8192 / BM), blk, 0, stream>>>(query, Wq, bq, Qh, 8192, 1024);
  gemm_wmma_k<float, float, _Float16>
      <<<dim3(NDIM / BN, 8192 / BM), blk, 0, stream>>>(key, Wk, bk, Kh, 8192, 1024);
  gemm_wmma_k<float, float, _Float16>
      <<<dim3(NDIM / BN, 8192 / BM), blk, 0, stream>>>(value, Wv, bv, Vh, 8192, 1024);

  // Softmax probabilities (f16), 512 q-tiles, 4 waves/block
  attn_softmax_k<<<dim3(128), dim3(128), 0, stream>>>(Qh, Kh, P);

  // O = P * V per batch: [2048 x 2048] x [2048 x 1024]
  for (int b = 0; b < 4; ++b)
    gemm_wmma_k<_Float16, _Float16, _Float16>
        <<<dim3(NDIM / BN, 2048 / BM), blk, 0, stream>>>(
            P + (size_t)b * 2048 * 2048, Vh + (size_t)b * 2048 * 1024, nullptr,
            Oh + (size_t)b * 2048 * 1024, 2048, 2048);

  // Final projection to fp32 output
  gemm_wmma_k<_Float16, float, float>
      <<<dim3(NDIM / BN, 8192 / BM), blk, 0, stream>>>(Oh, Wo, bo, out, 8192, 1024);
}
